// MultiHeadAttention_34634616275596
// MI455X (gfx1250) — compile-verified, matrix operand order checked by reading, not run on hardware
//
#include <hip/hip_runtime.h>

// ---------------------------------------------------------------------------
// MHA block for B=4, S=1024, D=1024, H=16, DH=64 on gfx1250 (MI455X).
// bf16 WMMA (v_wmma_f32_16x16x32_bf16) everywhere, flash-attention softmax.
// Faithful to the reference's raw (no-head-transpose) reshape:
//   Qv[b*16 + s/64][(s%64)*16 + h][dh] = Q[b][s][h*64+dh]
// ---------------------------------------------------------------------------

typedef __attribute__((ext_vector_type(16))) __bf16 v16bf;
typedef __attribute__((ext_vector_type(8)))  float  v8f;

union Frag {
  uint4 u[2];
  v16bf v;
};

#define WMMA_BF16(a, b, c)                                                   \
  __builtin_amdgcn_wmma_f32_16x16x32_bf16(false, (a), false, (b), (short)0,  \
                                          (c), false, false)

// Build a 16x32 bf16 A/B fragment: lane lo holds row lo, VGPR0-3 = K hi*8..,
// VGPR4-7 = K 16+hi*8..  ->  two 16B loads 32B apart.
static __device__ __forceinline__ void load_frag(Frag& f, const __bf16* base) {
  const uint4* p = reinterpret_cast<const uint4*>(base);
  f.u[0] = p[0];
  f.u[1] = p[2];
}

// ---------------------------------------------------------------------------
__global__ void cvt_f32_bf16(const float* __restrict__ in,
                             __bf16* __restrict__ out, int n) {
  int i = blockIdx.x * blockDim.x + threadIdx.x;
  if (i < n) out[i] = (__bf16)in[i];
}

// ---------------------------------------------------------------------------
// Fused Q/K/V projection: out = x @ W^T + b, written in attention layout.
// grid (N/128=8, M/128=32, 3), block 256 (8 waves, 2x4, wave tile 64Mx32N).
__global__ __launch_bounds__(256) void proj_qkv(
    const __bf16* __restrict__ X, const __bf16* __restrict__ Wq,
    const __bf16* __restrict__ Wk, const __bf16* __restrict__ Wv,
    const float* __restrict__ bq, const float* __restrict__ bk,
    const float* __restrict__ bv, __bf16* __restrict__ Qo,
    __bf16* __restrict__ Ko, __bf16* __restrict__ Vt) {
  const int mode = blockIdx.z;
  const __bf16* W = (mode == 0) ? Wq : (mode == 1) ? Wk : Wv;
  const float* bias = (mode == 0) ? bq : (mode == 1) ? bk : bv;

  const int lane = threadIdx.x & 31, wave = threadIdx.x >> 5;
  const int lo = lane & 15, hi = lane >> 4;
  const int m0 = blockIdx.y * 128 + (wave >> 2) * 64;
  const int n0 = blockIdx.x * 128 + (wave & 3) * 32;

  v8f acc[4][2] = {};
  for (int k0 = 0; k0 < 1024; k0 += 32) {
    Frag a[4], b[2];
#pragma unroll
    for (int i = 0; i < 4; i++)
      load_frag(a[i], X + (size_t)(m0 + i * 16 + lo) * 1024 + k0 + hi * 8);
#pragma unroll
    for (int j = 0; j < 2; j++)
      load_frag(b[j], W + (size_t)(n0 + j * 16 + lo) * 1024 + k0 + hi * 8);
#pragma unroll
    for (int i = 0; i < 4; i++)
#pragma unroll
      for (int j = 0; j < 2; j++)
        acc[i][j] = WMMA_BF16(a[i].v, b[j].v, acc[i][j]);
  }

#pragma unroll
  for (int i = 0; i < 4; i++)
#pragma unroll
    for (int j = 0; j < 2; j++)
#pragma unroll
      for (int r = 0; r < 8; r++) {
        int m = m0 + i * 16 + hi * 8 + r;     // token index b*1024+s
        int n = n0 + j * 16 + lo;             // feature d = h*64+dh
        float val = acc[i][j][r] + bias[n];
        int bb = m >> 10, s = m & 1023;
        int h = n >> 6, dh = n & 63;
        int bh = bb * 16 + (s >> 6);          // raw-view batch-head
        int sp = ((s & 63) << 4) + h;         // raw-view sequence pos
        if (mode == 2)
          Vt[((size_t)bh * 64 + dh) * 1024 + sp] = (__bf16)val;  // transposed
        else if (mode == 1)
          Ko[((size_t)bh * 1024 + sp) * 64 + dh] = (__bf16)val;
        else
          Qo[((size_t)bh * 1024 + sp) * 64 + dh] = (__bf16)val;
      }
}

// ---------------------------------------------------------------------------
// Flash attention over each [1024,64] slice. grid (S/128=8, BH=64), block 256.
// Each wave owns 16 query rows; key chunks of 64; online softmax.
__global__ __launch_bounds__(256) void flash_attn(
    const __bf16* __restrict__ Q, const __bf16* __restrict__ K,
    const __bf16* __restrict__ Vt, __bf16* __restrict__ Ctx) {
  __shared__ __align__(16) __bf16 lds_p[8][16 * 64];  // 2KB per wave
  const int lane = threadIdx.x & 31, wave = threadIdx.x >> 5;
  const int lo = lane & 15, hi = lane >> 4;
  const int bh = blockIdx.y;
  const int m0 = blockIdx.x * 128 + wave * 16;
  const __bf16* Qb = Q + (size_t)bh * 1024 * 64;
  const __bf16* Kb = K + (size_t)bh * 1024 * 64;
  const __bf16* Vb = Vt + (size_t)bh * 64 * 1024;

  Frag aq[2];
#pragma unroll
  for (int ks = 0; ks < 2; ks++)
    load_frag(aq[ks], Qb + (size_t)(m0 + lo) * 64 + ks * 32 + hi * 8);

  v8f O[4] = {};
  float mrow[8], lrow[8];
#pragma unroll
  for (int r = 0; r < 8; r++) { mrow[r] = -1e30f; lrow[r] = 0.f; }

  for (int kc = 0; kc < 1024; kc += 64) {
    // S = scale * Q @ K^T  (4 key tiles of 16)
    v8f st[4] = {};
#pragma unroll
    for (int t = 0; t < 4; t++)
#pragma unroll
      for (int ks = 0; ks < 2; ks++) {
        Frag bk;
        load_frag(bk, Kb + (size_t)(kc + t * 16 + lo) * 64 + ks * 32 + hi * 8);
        st[t] = WMMA_BF16(aq[ks].v, bk.v, st[t]);
      }
#pragma unroll
    for (int t = 0; t < 4; t++)
#pragma unroll
      for (int r = 0; r < 8; r++) st[t][r] = st[t][r] * 0.125f;

    // row max across 4 tiles + 16-lane groups (rows r / r+8 stay separate)
    float rmax[8], alpha[8], rsum[8];
#pragma unroll
    for (int r = 0; r < 8; r++) {
      float mx = fmaxf(fmaxf(st[0][r], st[1][r]), fmaxf(st[2][r], st[3][r]));
#pragma unroll
      for (int off = 8; off >= 1; off >>= 1)
        mx = fmaxf(mx, __shfl_xor(mx, off, 32));
      rmax[r] = mx;
    }
#pragma unroll
    for (int r = 0; r < 8; r++) {
      float mn = fmaxf(mrow[r], rmax[r]);
      alpha[r] = __expf(mrow[r] - mn);
      mrow[r] = mn;
      rsum[r] = 0.f;
    }
    // P = exp(S - m); stash bf16 P in wave-private LDS (C-layout -> A-layout)
#pragma unroll
    for (int t = 0; t < 4; t++)
#pragma unroll
      for (int r = 0; r < 8; r++) {
        float p = __expf(st[t][r] - mrow[r]);
        rsum[r] += p;
        lds_p[wave][(r + hi * 8) * 64 + t * 16 + lo] = (__bf16)p;
      }
#pragma unroll
    for (int r = 0; r < 8; r++) {
      float sres = rsum[r];
#pragma unroll
      for (int off = 8; off >= 1; off >>= 1) sres += __shfl_xor(sres, off, 32);
      lrow[r] = lrow[r] * alpha[r] + sres;
    }
#pragma unroll
    for (int t = 0; t < 4; t++)
#pragma unroll
      for (int r = 0; r < 8; r++) O[t][r] = O[t][r] * alpha[r];

    // wave-private LDS RAW: wait for the 32 ds_store_b16 above
    asm volatile("s_wait_dscnt 0x0" ::: "memory");

    Frag ap[2];
#pragma unroll
    for (int ks = 0; ks < 2; ks++) {
      const uint4* p = reinterpret_cast<const uint4*>(
          &lds_p[wave][lo * 64 + ks * 32 + hi * 8]);
      ap[ks].u[0] = p[0];
      ap[ks].u[1] = p[2];
    }
    // O += P @ V  (Vt rows are dh, contiguous along keys -> exact B frags)
#pragma unroll
    for (int t = 0; t < 4; t++)
#pragma unroll
      for (int ks = 0; ks < 2; ks++) {
        Frag bv;
        load_frag(bv, Vb + (size_t)(t * 16 + lo) * 1024 + kc + ks * 32 + hi * 8);
        O[t] = WMMA_BF16(ap[ks].v, bv.v, O[t]);
      }
  }

  // normalize and scatter back through the inverse raw-view map
#pragma unroll
  for (int r = 0; r < 8; r++) {
    float inv = 1.0f / lrow[r];
    int sp = m0 + r + hi * 8;
    int b = bh >> 4, a = bh & 15;
    int h = sp & 15, srem = sp >> 4;
    int s = a * 64 + srem;
#pragma unroll
    for (int t = 0; t < 4; t++) {
      int d = h * 64 + t * 16 + lo;
      Ctx[((size_t)(b * 1024 + s)) * 1024 + d] = (__bf16)(O[t][r] * inv);
    }
  }
}

// ---------------------------------------------------------------------------
// y = ctx @ Wo^T + bo + x  (fp32 out). grid (8, 32), block 256.
__global__ __launch_bounds__(256) void out_proj(
    const __bf16* __restrict__ Ctx, const __bf16* __restrict__ Wo,
    const float* __restrict__ bo, const float* __restrict__ X,
    float* __restrict__ Y) {
  const int lane = threadIdx.x & 31, wave = threadIdx.x >> 5;
  const int lo = lane & 15, hi = lane >> 4;
  const int m0 = blockIdx.y * 128 + (wave >> 2) * 64;
  const int n0 = blockIdx.x * 128 + (wave & 3) * 32;

  v8f acc[4][2] = {};
  for (int k0 = 0; k0 < 1024; k0 += 32) {
    Frag a[4], b[2];
#pragma unroll
    for (int i = 0; i < 4; i++)
      load_frag(a[i], Ctx + (size_t)(m0 + i * 16 + lo) * 1024 + k0 + hi * 8);
#pragma unroll
    for (int j = 0; j < 2; j++)
      load_frag(b[j], Wo + (size_t)(n0 + j * 16 + lo) * 1024 + k0 + hi * 8);
#pragma unroll
    for (int i = 0; i < 4; i++)
#pragma unroll
      for (int j = 0; j < 2; j++)
        acc[i][j] = WMMA_BF16(a[i].v, b[j].v, acc[i][j]);
  }
#pragma unroll
  for (int i = 0; i < 4; i++)
#pragma unroll
    for (int j = 0; j < 2; j++)
#pragma unroll
      for (int r = 0; r < 8; r++) {
        int m = m0 + i * 16 + hi * 8 + r;
        int n = n0 + j * 16 + lo;
        Y[(size_t)m * 1024 + n] =
            acc[i][j][r] + bo[n] + X[(size_t)m * 1024 + n];
      }
}

// ---------------------------------------------------------------------------
// Row LayerNorm over D=1024. One block per token, 256 threads x float4.
__global__ __launch_bounds__(256) void layernorm_k(
    const float* __restrict__ Y, const float* __restrict__ gamma,
    const float* __restrict__ beta, float* __restrict__ Out) {
  __shared__ float red[8];
  const int row = blockIdx.x;
  const int lane = threadIdx.x & 31, wave = threadIdx.x >> 5;
  const float* y = Y + (size_t)row * 1024;
  float4 v = reinterpret_cast<const float4*>(y)[threadIdx.x];

  float s = v.x + v.y + v.z + v.w;
#pragma unroll
  for (int off = 16; off >= 1; off >>= 1) s += __shfl_xor(s, off, 32);
  if (lane == 0) red[wave] = s;
  __syncthreads();
  float tot = 0.f;
#pragma unroll
  for (int w = 0; w < 8; w++) tot += red[w];
  const float mu = tot * (1.0f / 1024.0f);
  __syncthreads();

  float dx = v.x - mu, dy = v.y - mu, dz = v.z - mu, dw = v.w - mu;
  float s2 = dx * dx + dy * dy + dz * dz + dw * dw;
#pragma unroll
  for (int off = 16; off >= 1; off >>= 1) s2 += __shfl_xor(s2, off, 32);
  if (lane == 0) red[wave] = s2;
  __syncthreads();
  float tot2 = 0.f;
#pragma unroll
  for (int w = 0; w < 8; w++) tot2 += red[w];
  const float rstd = rsqrtf(tot2 * (1.0f / 1024.0f) + 1e-5f);

  float4 g = reinterpret_cast<const float4*>(gamma)[threadIdx.x];
  float4 bt = reinterpret_cast<const float4*>(beta)[threadIdx.x];
  float4 o;
  o.x = dx * rstd * g.x + bt.x;
  o.y = dy * rstd * g.y + bt.y;
  o.z = dz * rstd * g.z + bt.z;
  o.w = dw * rstd * g.w + bt.w;
  reinterpret_cast<float4*>(Out + (size_t)row * 1024)[threadIdx.x] = o;
}

// ---------------------------------------------------------------------------
extern "C" void kernel_launch(void* const* d_in, const int* in_sizes, int n_in,
                              void* d_out, int out_size, void* d_ws,
                              size_t ws_size, hipStream_t stream) {
  (void)in_sizes; (void)n_in; (void)out_size; (void)ws_size;
  const float* x  = (const float*)d_in[0];
  const float* Wq = (const float*)d_in[1];
  const float* bq = (const float*)d_in[2];
  const float* Wk = (const float*)d_in[3];
  const float* bk = (const float*)d_in[4];
  const float* Wv = (const float*)d_in[5];
  const float* bv = (const float*)d_in[6];
  const float* Wo = (const float*)d_in[7];
  const float* bo = (const float*)d_in[8];
  const float* gamma = (const float*)d_in[9];
  const float* beta  = (const float*)d_in[10];
  float* out = (float*)d_out;

  char* ws = (char*)d_ws;
  const size_t MB = 1024 * 1024;
  __bf16* xb  = (__bf16*)(ws + 0 * MB);   // 8 MB   x bf16 [4096,1024]
  __bf16* wqb = (__bf16*)(ws + 8 * MB);   // 2 MB
  __bf16* wkb = (__bf16*)(ws + 10 * MB);  // 2 MB
  __bf16* wvb = (__bf16*)(ws + 12 * MB);  // 2 MB
  __bf16* wob = (__bf16*)(ws + 14 * MB);  // 2 MB
  __bf16* Qb  = (__bf16*)(ws + 16 * MB);  // 8 MB  [64][1024][64]
  __bf16* Kb  = (__bf16*)(ws + 24 * MB);  // 8 MB  [64][1024][64]
  __bf16* Vtb = (__bf16*)(ws + 32 * MB);  // 8 MB  [64][64][1024] (transposed)
  __bf16* ctx = (__bf16*)(ws + 40 * MB);  // 8 MB  [4096][1024]
  float*  y   = (float*)(ws + 48 * MB);   // 16 MB [4096][1024]

  cvt_f32_bf16<<<(4096 * 1024) / 256, 256, 0, stream>>>(x, xb, 4096 * 1024);
  cvt_f32_bf16<<<(1024 * 1024) / 256, 256, 0, stream>>>(Wq, wqb, 1024 * 1024);
  cvt_f32_bf16<<<(1024 * 1024) / 256, 256, 0, stream>>>(Wk, wkb, 1024 * 1024);
  cvt_f32_bf16<<<(1024 * 1024) / 256, 256, 0, stream>>>(Wv, wvb, 1024 * 1024);
  cvt_f32_bf16<<<(1024 * 1024) / 256, 256, 0, stream>>>(Wo, wob, 1024 * 1024);

  proj_qkv<<<dim3(8, 32, 3), 256, 0, stream>>>(xb, wqb, wkb, wvb, bq, bk, bv,
                                               Qb, Kb, Vtb);
  flash_attn<<<dim3(8, 64), 256, 0, stream>>>(Qb, Kb, Vtb, ctx);
  out_proj<<<dim3(8, 32), 256, 0, stream>>>(ctx, wob, bo, x, y);
  layernorm_k<<<4096, 256, 0, stream>>>(y, gamma, beta, out);
}
